// HtrgGraphAttentionLayer_80668075753787
// MI455X (gfx1250) — matrix operand order repaired and checked by reading, not compiled, not run on hardware
//
#include <hip/hip_runtime.h>
#include <math.h>

typedef __attribute__((ext_vector_type(16))) __bf16 v16bf;
typedef __attribute__((ext_vector_type(8)))  __bf16 v8bf;
typedef __attribute__((ext_vector_type(8)))  float  v8f;
typedef __attribute__((ext_vector_type(4)))  unsigned int v4u;

#define NB   2
#define NN   1024
#define N1   512
#define DD   64

// workspace byte offsets
#define WS_X      0u          // fp32 x        : 2*1024*64*4 = 524288
#define WS_XB     524288u     // bf16 x        : 262144
#define WS_XBT    786432u     // bf16 x^T      : 262144
#define WS_WATTT  1048576u    // bf16 Watt^T   : 8192
#define WS_SCORES 1056768u    // fp32 scores   : 8388608
#define WS_AGG    9445376u    // fp32 agg      : 524288
#define WS_SMAST  9969664u    // fp32 master s : 8192

// hardware tanh (TRANS32 op on gfx1250) instead of branchy ocml tanhf
__device__ __forceinline__ float fast_tanh(float x) {
#if __has_builtin(__builtin_amdgcn_tanhf)
    return __builtin_amdgcn_tanhf(x);
#else
    float y;
    asm("v_tanh_f32 %0, %1" : "=v"(y) : "v"(x));
    return y;
#endif
}

// 8 packed bf16 multiplies via V_PK_MUL_BF16 (2 bf16 per dword)
__device__ __forceinline__ v4u pk_mul_bf16x8(v4u a, v4u b) {
    unsigned int r0, r1, r2, r3;
    asm("v_pk_mul_bf16 %0, %1, %2" : "=v"(r0) : "v"(a.x), "v"(b.x));
    asm("v_pk_mul_bf16 %0, %1, %2" : "=v"(r1) : "v"(a.y), "v"(b.y));
    asm("v_pk_mul_bf16 %0, %1, %2" : "=v"(r2) : "v"(a.z), "v"(b.z));
    asm("v_pk_mul_bf16 %0, %1, %2" : "=v"(r3) : "v"(a.w), "v"(b.w));
    v4u r = {r0, r1, r2, r3};
    return r;
}

// ---- transpose Watt to bf16 (B-operand wants contiguous K per output col) ----
__global__ void k_wattT(const float* __restrict__ Watt, __bf16* __restrict__ wattT) {
    int t = blockIdx.x * blockDim.x + threadIdx.x;   // 4096 threads
    int k = t >> 6, n = t & 63;
    wattT[n * 64 + k] = (__bf16)Watt[k * 64 + n];
}

// ---- projections: x = [x1@Wt1+bt1 ; x2@Wt2+bt2]; emit fp32, bf16, bf16^T ----
__global__ void k_proj(const float* __restrict__ x1, const float* __restrict__ x2,
                       const float* __restrict__ Wt1, const float* __restrict__ bt1,
                       const float* __restrict__ Wt2, const float* __restrict__ bt2,
                       float* __restrict__ x, __bf16* __restrict__ xb,
                       __bf16* __restrict__ xbT) {
    int row = blockIdx.x;            // 0..2047
    int b = row >> 10, n = row & 1023;
    int d = threadIdx.x;             // 0..63
    const float *W, *bias, *src;
    if (n < N1) { W = Wt1; bias = bt1; src = x1 + (size_t)(b * N1 + n) * DD; }
    else        { W = Wt2; bias = bt2; src = x2 + (size_t)(b * N1 + (n - N1)) * DD; }
    __shared__ float srow[DD];
    srow[d] = src[d];
    __syncthreads();
    float acc = bias[d];
#pragma unroll
    for (int k = 0; k < DD; ++k) acc = fmaf(srow[k], W[k * DD + d], acc);
    x[(size_t)row * DD + d]  = acc;
    xb[(size_t)row * DD + d] = (__bf16)acc;
    xbT[((size_t)b * DD + d) * NN + n] = (__bf16)acc;
}

// ---- pairwise scores: per wave one (b,i); tile j by 16; bf16 WMMA ----
__global__ __launch_bounds__(256)
void k_pair(const __bf16* __restrict__ xb, const __bf16* __restrict__ wattT,
            const float* __restrict__ batt, const float* __restrict__ w11,
            const float* __restrict__ w22, const float* __restrict__ w12,
            float* __restrict__ scores) {
    int wid  = blockIdx.x * 8 + (threadIdx.x >> 5);
    int lane = threadIdx.x & 31;
    int half = lane >> 4, col = lane & 15;
    int b = wid >> 10, i = wid & 1023;

    // B fragments (Watt^T), constant over the whole j loop: 4 N-tiles x 2 K-chunks.
    v16bf Bf[4][2];
#pragma unroll
    for (int nt = 0; nt < 4; ++nt)
#pragma unroll
        for (int kc = 0; kc < 2; ++kc)
            Bf[nt][kc] = *(const v16bf*)(wattT + (nt * 16 + col) * 64 + kc * 32 + half * 16);

    // x_i slices for this lane's K ranges (packed bf16 pairs)
    const __bf16* xi = xb + (size_t)(b * NN + i) * DD;
    v4u xiP[2][2];
#pragma unroll
    for (int kc = 0; kc < 2; ++kc)
#pragma unroll
        for (int g = 0; g < 2; ++g)
            xiP[kc][g] = *(const v4u*)(xi + kc * 32 + g * 8 + half * 16);

    float battv[4], wLv[4], wRv[4];
    const float* wL = (i < N1) ? w11 : w12;
    const float* wR = (i < N1) ? w12 : w22;
#pragma unroll
    for (int nt = 0; nt < 4; ++nt) {
        battv[nt] = batt[nt * 16 + col];
        wLv[nt]   = wL[nt * 16 + col];
        wRv[nt]   = wR[nt * 16 + col];
    }

    float* srow = scores + ((size_t)(b * NN + i)) * NN;

#pragma unroll 2
    for (int j0 = 0; j0 < NN; j0 += 16) {
        const __bf16* xj = xb + (size_t)(b * NN + j0 + col) * DD;  // row m = col
        // prefetch next tile's row slice while this tile computes
        if (j0 + 16 < NN) __builtin_prefetch(xj + 16 * DD, 0, 3);
        v16bf Af[2];
#pragma unroll
        for (int kc = 0; kc < 2; ++kc) {
            v4u lo = pk_mul_bf16x8(*(const v4u*)(xj + kc * 32 + half * 16),      xiP[kc][0]);
            v4u hi = pk_mul_bf16x8(*(const v4u*)(xj + kc * 32 + 8 + half * 16), xiP[kc][1]);
            union { v4u u[2]; v16bf v; } cu;
            cu.u[0] = lo; cu.u[1] = hi;
            Af[kc] = cu.v;
        }
        bool leftj = (j0 < N1);
        float part[8];
#pragma unroll
        for (int r = 0; r < 8; ++r) part[r] = 0.f;
#pragma unroll
        for (int nt = 0; nt < 4; ++nt) {
            v8f c = {};
            c = __builtin_amdgcn_wmma_f32_16x16x32_bf16(false, Af[0], false, Bf[nt][0],
                                                        (short)0, c, false, false);
            c = __builtin_amdgcn_wmma_f32_16x16x32_bf16(false, Af[1], false, Bf[nt][1],
                                                        (short)0, c, false, false);
            float ws = leftj ? wLv[nt] : wRv[nt];
            float bb = battv[nt];
#pragma unroll
            for (int r = 0; r < 8; ++r)
                part[r] += fast_tanh(c[r] + bb) * ws;
        }
        // sum over the 16 n-lanes within each half (xor of low 4 bits keeps halves apart)
#pragma unroll
        for (int off = 1; off < 16; off <<= 1)
#pragma unroll
            for (int r = 0; r < 8; ++r)
                part[r] += __shfl_xor(part[r], off, 32);
        if (col == 0) {
            float* o = srow + j0 + half * 8;   // lane0 -> rows 0..7, lane16 -> rows 8..15
#pragma unroll
            for (int r = 0; r < 8; ++r) o[r] = part[r];
        }
    }
}

// ---- row softmax over j (in place) ----
__global__ __launch_bounds__(256)
void k_softmax(float* __restrict__ scores) {
    float* s = scores + (size_t)blockIdx.x * NN;
    int t = threadIdx.x;
    __shared__ float red[256];
    float v[4], mx = -3.4e38f;
#pragma unroll
    for (int q = 0; q < 4; ++q) { v[q] = s[t + q * 256]; mx = fmaxf(mx, v[q]); }
    red[t] = mx; __syncthreads();
    for (int o = 128; o > 0; o >>= 1) { if (t < o) red[t] = fmaxf(red[t], red[t + o]); __syncthreads(); }
    mx = red[0]; __syncthreads();
    float sum = 0.f;
#pragma unroll
    for (int q = 0; q < 4; ++q) { v[q] = __expf(v[q] - mx); sum += v[q]; }
    red[t] = sum; __syncthreads();
    for (int o = 128; o > 0; o >>= 1) { if (t < o) red[t] += red[t + o]; __syncthreads(); }
    float inv = 1.f / red[0];
#pragma unroll
    for (int q = 0; q < 4; ++q) s[t + q * 256] = v[q] * inv;
}

// ---- agg = att @ x  (per wave: one 16-row i tile, all 64 cols) ----
__global__ __launch_bounds__(256)
void k_agg(const float* __restrict__ att, const __bf16* __restrict__ xbT,
           float* __restrict__ agg) {
    int wid  = blockIdx.x * 8 + (threadIdx.x >> 5);   // 0..127
    int lane = threadIdx.x & 31;
    int half = lane >> 4, col = lane & 15;
    int b = wid >> 6, it = wid & 63;
    v8f c[4] = {};
    const float*  arow = att + ((size_t)(b * NN + it * 16 + col)) * NN;  // row m = col
    const __bf16* xt   = xbT + (size_t)b * DD * NN;
    for (int kc = 0; kc < NN; kc += 32) {
        v16bf Af;
        const float* a0 = arow + kc + half * 16;
        const float* a1 = arow + kc + 8 + half * 16;
#pragma unroll
        for (int e = 0; e < 8; ++e) { Af[e] = (__bf16)a0[e]; Af[8 + e] = (__bf16)a1[e]; }
#pragma unroll
        for (int nt = 0; nt < 4; ++nt) {
            v16bf Bf = *(const v16bf*)(xt + (nt * 16 + col) * NN + kc + half * 16);
            c[nt] = __builtin_amdgcn_wmma_f32_16x16x32_bf16(false, Af, false, Bf,
                                                            (short)0, c[nt], false, false);
        }
    }
#pragma unroll
    for (int nt = 0; nt < 4; ++nt)
#pragma unroll
        for (int r = 0; r < 8; ++r)
            agg[((size_t)(b * NN + it * 16 + r + 8 * half)) * DD + nt * 16 + col] = c[nt][r];
}

// ---- master attention logits: s[b,n] = tanh((x_n * m) @ WattM + battM) . wM ----
__global__ void k_master1(const float* __restrict__ x, const float* __restrict__ master,
                          const float* __restrict__ WattM, const float* __restrict__ battM,
                          const float* __restrict__ wM, float* __restrict__ s) {
    int t = blockIdx.x * blockDim.x + threadIdx.x;   // 0..2047
    int b = t >> 10;
    const float* xr = x + (size_t)t * DD;
    const float* mr = master + b * DD;
    float xm[DD];
#pragma unroll
    for (int k = 0; k < DD; ++k) xm[k] = xr[k] * mr[k];
    float acc_s = 0.f;
    for (int dc = 0; dc < DD; ++dc) {
        float acc = battM[dc];
#pragma unroll
        for (int k = 0; k < DD; ++k) acc = fmaf(xm[k], WattM[k * DD + dc], acc);
        acc_s += fast_tanh(acc) * wM[dc];
    }
    s[t] = acc_s;
}

// ---- master softmax + aggregate + output projection ----
__global__ __launch_bounds__(256)
void k_master2(const float* __restrict__ s, const float* __restrict__ x,
               const float* __restrict__ master, const float* __restrict__ WwaM,
               const float* __restrict__ bwaM, const float* __restrict__ WwoaM,
               const float* __restrict__ bwoaM, float* __restrict__ out_master) {
    int b = blockIdx.x, t = threadIdx.x;
    __shared__ float p[NN];
    __shared__ float red[256];
    __shared__ float magg[DD];
    float v[4], mx = -3.4e38f;
#pragma unroll
    for (int q = 0; q < 4; ++q) { v[q] = s[b * NN + t + q * 256]; mx = fmaxf(mx, v[q]); }
    red[t] = mx; __syncthreads();
    for (int o = 128; o > 0; o >>= 1) { if (t < o) red[t] = fmaxf(red[t], red[t + o]); __syncthreads(); }
    mx = red[0]; __syncthreads();
    float sum = 0.f;
#pragma unroll
    for (int q = 0; q < 4; ++q) { v[q] = __expf(v[q] - mx); sum += v[q]; }
    red[t] = sum; __syncthreads();
    for (int o = 128; o > 0; o >>= 1) { if (t < o) red[t] += red[t + o]; __syncthreads(); }
    float inv = 1.f / red[0]; __syncthreads();
#pragma unroll
    for (int q = 0; q < 4; ++q) p[t + q * 256] = v[q] * inv;
    __syncthreads();
    if (t < DD) {
        float acc = 0.f;
        for (int n = 0; n < NN; ++n) acc = fmaf(p[n], x[((size_t)b * NN + n) * DD + t], acc);
        magg[t] = acc;
    }
    __syncthreads();
    if (t < DD) {
        float acc = bwaM[t] + bwoaM[t];
#pragma unroll
        for (int k = 0; k < DD; ++k) {
            acc = fmaf(magg[k], WwaM[k * DD + t], acc);
            acc = fmaf(master[b * DD + k], WwoaM[k * DD + t], acc);
        }
        out_master[b * DD + t] = acc;
    }
}

// ---- xo = agg@Wwa + x@Wwoa + biases -> BN(eval) -> SELU -> split outputs ----
__global__ void k_final(const float* __restrict__ agg, const float* __restrict__ x,
                        const float* __restrict__ Wwa, const float* __restrict__ bwa,
                        const float* __restrict__ Wwoa, const float* __restrict__ bwoa,
                        const float* __restrict__ gamma, const float* __restrict__ beta,
                        const float* __restrict__ rm, const float* __restrict__ rv,
                        float* __restrict__ out) {
    int row = blockIdx.x;                 // 0..2047
    int b = row >> 10, n = row & 1023;
    int d = threadIdx.x;                  // 0..63
    __shared__ float ar[DD], xr[DD];
    ar[d] = agg[(size_t)row * DD + d];
    xr[d] = x[(size_t)row * DD + d];
    __syncthreads();
    float acc = bwa[d] + bwoa[d];
#pragma unroll
    for (int k = 0; k < DD; ++k) {
        acc = fmaf(ar[k], Wwa[k * DD + d], acc);
        acc = fmaf(xr[k], Wwoa[k * DD + d], acc);
    }
    acc = (acc - rm[d]) * rsqrtf(rv[d] + 1e-5f) * gamma[d] + beta[d];
    const float alpha = 1.6732632423543772f, scale = 1.0507009873554805f;
    acc = scale * (acc > 0.f ? acc : alpha * (__expf(acc) - 1.f));
    float* dst = (n < N1) ? (out + ((size_t)b * N1 + n) * DD + d)
                          : (out + (size_t)NB * N1 * DD + ((size_t)b * N1 + (n - N1)) * DD + d);
    *dst = acc;
}

extern "C" void kernel_launch(void* const* d_in, const int* in_sizes, int n_in,
                              void* d_out, int out_size, void* d_ws, size_t ws_size,
                              hipStream_t stream) {
    const float* x1     = (const float*)d_in[0];
    const float* x2     = (const float*)d_in[1];
    const float* master = (const float*)d_in[2];
    const float* Wt1    = (const float*)d_in[3];
    const float* bt1    = (const float*)d_in[4];
    const float* Wt2    = (const float*)d_in[5];
    const float* bt2    = (const float*)d_in[6];
    const float* Watt   = (const float*)d_in[7];
    const float* batt   = (const float*)d_in[8];
    const float* WattM  = (const float*)d_in[9];
    const float* battM  = (const float*)d_in[10];
    const float* w11    = (const float*)d_in[11];
    const float* w22    = (const float*)d_in[12];
    const float* w12    = (const float*)d_in[13];
    const float* wM     = (const float*)d_in[14];
    const float* Wwa    = (const float*)d_in[15];
    const float* bwa    = (const float*)d_in[16];
    const float* Wwoa   = (const float*)d_in[17];
    const float* bwoa   = (const float*)d_in[18];
    const float* WwaM   = (const float*)d_in[19];
    const float* bwaM   = (const float*)d_in[20];
    const float* WwoaM  = (const float*)d_in[21];
    const float* bwoaM  = (const float*)d_in[22];
    const float* gamma  = (const float*)d_in[23];
    const float* beta   = (const float*)d_in[24];
    const float* rm     = (const float*)d_in[25];
    const float* rv     = (const float*)d_in[26];

    char* ws = (char*)d_ws;
    float*  x      = (float*)(ws + WS_X);
    __bf16* xb     = (__bf16*)(ws + WS_XB);
    __bf16* xbT    = (__bf16*)(ws + WS_XBT);
    __bf16* wattT  = (__bf16*)(ws + WS_WATTT);
    float*  scores = (float*)(ws + WS_SCORES);
    float*  agg    = (float*)(ws + WS_AGG);
    float*  smast  = (float*)(ws + WS_SMAST);

    k_wattT  <<<16,   256, 0, stream>>>(Watt, wattT);
    k_proj   <<<2048,  64, 0, stream>>>(x1, x2, Wt1, bt1, Wt2, bt2, x, xb, xbT);
    k_pair   <<<256,  256, 0, stream>>>(xb, wattT, batt, w11, w22, w12, scores);
    k_softmax<<<2048, 256, 0, stream>>>(scores);
    k_agg    <<<16,   256, 0, stream>>>(scores, xbT, agg);
    k_master1<<<8,    256, 0, stream>>>(x, master, WattM, battM, wM, smast);
    k_master2<<<2,    256, 0, stream>>>(smast, x, master, WwaM, bwaM, WwoaM, bwoaM,
                                        (float*)d_out + (size_t)2 * NB * N1 * DD);
    k_final  <<<2048,  64, 0, stream>>>(agg, x, Wwa, bwa, Wwoa, bwoa, gamma, beta, rm, rv,
                                        (float*)d_out);
}
